// GCN_3023656976550
// MI455X (gfx1250) — matrix operand order repaired and checked by reading, not compiled, not run on hardware
//
#include <hip/hip_runtime.h>
#include <hip/hip_fp16.h>

typedef __attribute__((ext_vector_type(16))) _Float16 v16h;
typedef __attribute__((ext_vector_type(8)))  _Float16 v8h;
typedef __attribute__((ext_vector_type(8)))  float    v8f;
typedef unsigned int u32x4 __attribute__((ext_vector_type(4)));
typedef int          i32x4 __attribute__((ext_vector_type(4)));
typedef int          i32x8 __attribute__((ext_vector_type(8)));

#define NTHREADS 256

// ---------------------------------------------------------------------------
// Degree / normalization:  deg[i] = 1 (self loop) + #incoming edges
// then dinv[i] = rsqrt(deg[i])  (deg >= 1 always, so no zero guard needed)
// ---------------------------------------------------------------------------
__global__ void deg_init_k(float* __restrict__ deg, int n) {
  int i = blockIdx.x * blockDim.x + threadIdx.x;
  if (i < n) deg[i] = 1.0f;
}

__global__ void deg_count_k(const int* __restrict__ dst, float* __restrict__ deg, int e) {
  int i = blockIdx.x * blockDim.x + threadIdx.x;
  if (i < e) atomicAdd(&deg[dst[i]], 1.0f);
}

__global__ void deg_rsqrt_k(float* __restrict__ deg, int n) {
  int i = blockIdx.x * blockDim.x + threadIdx.x;
  if (i < n) deg[i] = rsqrtf(deg[i]);
}

// ---------------------------------------------------------------------------
// Self-loop term: acc[i,:] = T[i,:] * dinv[i]^2    (F = 128, float4 per thread)
// ---------------------------------------------------------------------------
__global__ void selfloop_init_k(const float* __restrict__ T,
                                const float* __restrict__ dinv,
                                float* __restrict__ acc, int n) {
  int i = blockIdx.x * blockDim.x + threadIdx.x;      // over n*32 float4 slots
  if (i >= n * 32) return;
  int node = i >> 5;
  float s = dinv[node];
  s = s * s;
  float4 v = ((const float4*)T)[i];
  ((float4*)acc)[i] = make_float4(v.x * s, v.y * s, v.z * s, v.w * s);
}

// ---------------------------------------------------------------------------
// Edge scatter: one wave per edge, float4 per lane (32 lanes * 4 = 128 feats).
// acc[dst,:] += T[src,:] * dinv[src]*dinv[dst]   via global f32 atomics.
// edge index is wave-uniform -> force scalar loads via readfirstlane.
// ---------------------------------------------------------------------------
__global__ void scatter_k(const float* __restrict__ T,
                          const float* __restrict__ dinv,
                          const int* __restrict__ src,
                          const int* __restrict__ dst,
                          float* __restrict__ acc, int e) {
  int edge = blockIdx.x * (NTHREADS / 32) + (threadIdx.x >> 5);
  if (edge >= e) return;
  edge = __builtin_amdgcn_readfirstlane(edge);
  int lane = threadIdx.x & 31;
  int s = src[edge];
  int d = dst[edge];
  float w = dinv[s] * dinv[d];
  float4 v = ((const float4*)(T + (size_t)s * 128))[lane];
  float* ap = acc + (size_t)d * 128 + lane * 4;
  atomicAdd(ap + 0, v.x * w);
  atomicAdd(ap + 1, v.y * w);
  atomicAdd(ap + 2, v.z * w);
  atomicAdd(ap + 3, v.w * w);
}

// ---------------------------------------------------------------------------
// WMMA GEMM:  Out[M,NOUT] = post( pre(X[M,K]) @ W[K,NOUT] )
//   PREBIAS : x <- relu(x + preBias[k])  (fused previous-layer epilogue)
//   postBias/POSTRELU : y <- [relu](y + postBias[n])
// One wave computes a 16-row stripe across all NOUT columns.
// 256 threads = 8 waves = 128 rows per block. W^T staged in LDS as f16.
// Weight staging uses the Tensor Data Mover when the fp32 staging buffer
// fits alongside the f16 buffer; otherwise direct VGPR-path staging.
// ---------------------------------------------------------------------------
template<int K, int NOUT, bool PREBIAS, bool POSTRELU>
__global__ __launch_bounds__(NTHREADS) void gemm_wmma_k(
    const float* __restrict__ X, const float* __restrict__ W,
    const float* __restrict__ preBias, const float* __restrict__ postBias,
    float* __restrict__ Out, int M) {
  constexpr int KP = K + 8;           // padded f16 stride: 16B align, no bank conflicts
  constexpr int NT = NOUT / 16;
  constexpr bool USE_TDM =
      ((size_t)K * NOUT * 4 + (size_t)NOUT * KP * 2) <= (size_t)(150 * 1024);
  __shared__ __align__(16) _Float16 ldsW[NOUT * KP];   // W^T as f16

  if constexpr (USE_TDM) {
    // --- TDM: DMA the whole K x NOUT fp32 weight tile into LDS ---
    __shared__ __align__(16) float ldsW32[K * NOUT];
    if (threadIdx.x < 32) {
      unsigned long long ga = (unsigned long long)(uintptr_t)W;
      unsigned ldsOff = (unsigned)(uintptr_t)&ldsW32[0];
      u32x4 g0;
      g0[0] = 1u;                                        // count=1, user mode
      g0[1] = ldsOff;                                    // lds_addr (bytes)
      g0[2] = (unsigned)(ga & 0xffffffffu);              // global_addr[31:0]
      g0[3] = (unsigned)((ga >> 32) & 0x01ffffffu)       // global_addr[56:32]
              | (2u << 30);                              // type = 2 ("image")
      i32x8 g1;
      g1[0] = (int)(2u << 16);                           // data_size = 4 bytes
      g1[1] = (int)((unsigned)NOUT << 16);               // tensor_dim0[15:0]
      g1[2] = (int)((((unsigned)NOUT >> 16) & 0xffffu)   // tensor_dim0[31:16]
              | ((unsigned)K << 16));                    // tensor_dim1[15:0]
      g1[3] = (int)((((unsigned)K >> 16) & 0xffffu)      // tensor_dim1[31:16]
              | ((unsigned)NOUT << 16));                 // tile_dim0 = NOUT
      g1[4] = (int)(unsigned)K;                          // tile_dim1 = K, tile_dim2 = 0
      g1[5] = (int)(unsigned)NOUT;                       // tensor_dim0_stride[31:0]
      g1[6] = 0;                                         // stride hi / dim1_stride lo
      g1[7] = 0;
      i32x4 gz4;
      gz4[0] = 0; gz4[1] = 0; gz4[2] = 0; gz4[3] = 0;
      i32x8 gz8;
      gz8[0] = 0; gz8[1] = 0; gz8[2] = 0; gz8[3] = 0;
      gz8[4] = 0; gz8[5] = 0; gz8[6] = 0; gz8[7] = 0;
      // amdgpu-toolchain (clang-23) 6-arg form:
      // (u32x4 g0, i32x8 g1, i32x4 g2, i32x4 g3, i32x8 extra, i32 cpol)
      __builtin_amdgcn_tensor_load_to_lds(g0, g1, gz4, gz4, gz8, 0);
      __builtin_amdgcn_s_wait_tensorcnt(0);
    }
    __syncthreads();
    // LDS fp32 -> LDS f16 transpose
    for (int i = threadIdx.x; i < K * NOUT; i += NTHREADS) {
      int k = i / NOUT;
      int n = i - k * NOUT;
      ldsW[n * KP + k] = (_Float16)ldsW32[i];
    }
    __syncthreads();
  } else {
    // --- direct staging (coalesced global read, strided LDS write) ---
    for (int i = threadIdx.x; i < K * NOUT; i += NTHREADS) {
      int k = i / NOUT;
      int n = i - k * NOUT;
      ldsW[n * KP + k] = (_Float16)W[i];
    }
    __syncthreads();
  }

  const int lane = threadIdx.x & 31;
  const int wave = threadIdx.x >> 5;
  const int kh   = lane >> 4;        // which K-half this lane holds
  const int l15  = lane & 15;

  const int rowBase = blockIdx.x * 128 + wave * 16;

  v8f acc[NT];
#pragma unroll
  for (int t = 0; t < NT; ++t)
#pragma unroll
    for (int j = 0; j < 8; ++j) acc[t][j] = 0.0f;

  // A-matrix row for this lane (clamped for the tail block; stores are guarded,
  // and clamping keeps EXEC all-ones as WMMA requires).
  int arow = rowBase + l15;
  if (arow >= M) arow = M - 1;
  const float* __restrict__ xrow = X + (size_t)arow * K;

#pragma unroll
  for (int k0 = 0; k0 < K; k0 += 32) {
    // ---- A fragment: 16x32 f16. ISA layout: element j -> K = kh*8 + (j&7) + (j>>3)*16
    const int kb = k0 + kh * 8;
    const float4* __restrict__ xv = (const float4*)(xrow + kb);
    float4 p0 = xv[0], p1 = xv[1];   // k = kb .. kb+7
    float4 q0 = xv[4], q1 = xv[5];   // k = kb+16 .. kb+23
    if constexpr (PREBIAS) {
      const float4* __restrict__ bv = (const float4*)(preBias + kb);
      float4 c0 = bv[0], c1 = bv[1], d0 = bv[4], d1 = bv[5];
      p0 = make_float4(fmaxf(p0.x + c0.x, 0.f), fmaxf(p0.y + c0.y, 0.f),
                       fmaxf(p0.z + c0.z, 0.f), fmaxf(p0.w + c0.w, 0.f));
      p1 = make_float4(fmaxf(p1.x + c1.x, 0.f), fmaxf(p1.y + c1.y, 0.f),
                       fmaxf(p1.z + c1.z, 0.f), fmaxf(p1.w + c1.w, 0.f));
      q0 = make_float4(fmaxf(q0.x + d0.x, 0.f), fmaxf(q0.y + d0.y, 0.f),
                       fmaxf(q0.z + d0.z, 0.f), fmaxf(q0.w + d0.w, 0.f));
      q1 = make_float4(fmaxf(q1.x + d1.x, 0.f), fmaxf(q1.y + d1.y, 0.f),
                       fmaxf(q1.z + d1.z, 0.f), fmaxf(q1.w + d1.w, 0.f));
    }
    v16h a;
    a[0]  = (_Float16)p0.x; a[1]  = (_Float16)p0.y;
    a[2]  = (_Float16)p0.z; a[3]  = (_Float16)p0.w;
    a[4]  = (_Float16)p1.x; a[5]  = (_Float16)p1.y;
    a[6]  = (_Float16)p1.z; a[7]  = (_Float16)p1.w;
    a[8]  = (_Float16)q0.x; a[9]  = (_Float16)q0.y;
    a[10] = (_Float16)q0.z; a[11] = (_Float16)q0.w;
    a[12] = (_Float16)q1.x; a[13] = (_Float16)q1.y;
    a[14] = (_Float16)q1.z; a[15] = (_Float16)q1.w;
    if (k0 + 32 < K) __builtin_prefetch(xrow + k0 + 32, 0, 0);

    // ---- B fragments from LDS + WMMA over all N tiles.
    // B layout: lane = column, element j -> K = kh*8 + (j&7) + (j>>3)*16 (matches A).
#pragma unroll
    for (int t = 0; t < NT; ++t) {
      const _Float16* bp = &ldsW[(t * 16 + l15) * KP + k0 + kh * 8];
      v8h b0 = *(const v8h*)bp;
      v8h b1 = *(const v8h*)(bp + 16);
      v16h b;
#pragma unroll
      for (int j = 0; j < 8; ++j) { b[j] = b0[j]; b[j + 8] = b1[j]; }
      acc[t] = __builtin_amdgcn_wmma_f32_16x16x32_f16(
          false, a, false, b, (short)0, acc[t], false, false);
    }
  }

  // ---- Epilogue. C layout: lane = column, VGPR r -> row M = kh*8 + r.
  const int rbase = kh * 8;
#pragma unroll
  for (int t = 0; t < NT; ++t) {
    int n = t * 16 + l15;
    float bv = postBias ? postBias[n] : 0.0f;
#pragma unroll
    for (int r = 0; r < 8; ++r) {
      int m = rowBase + rbase + r;
      if (m < M) {
        float v = acc[t][r] + bv;
        if (POSTRELU) v = fmaxf(v, 0.0f);
        Out[(size_t)m * NOUT + n] = v;
      }
    }
  }
}

// ---------------------------------------------------------------------------
// Host-side orchestration
// inputs: x, edge_index, W1, b1, W2, b2, Wf1, bf1, Wf2, bf2
// ---------------------------------------------------------------------------
extern "C" void kernel_launch(void* const* d_in, const int* in_sizes, int n_in,
                              void* d_out, int out_size, void* d_ws, size_t ws_size,
                              hipStream_t stream) {
  const float* x   = (const float*)d_in[0];
  const int*   ei  = (const int*)d_in[1];
  const float* W1  = (const float*)d_in[2];
  const float* b1  = (const float*)d_in[3];
  const float* W2  = (const float*)d_in[4];
  const float* b2  = (const float*)d_in[5];
  const float* Wf1 = (const float*)d_in[6];
  const float* bf1 = (const float*)d_in[7];
  const float* Wf2 = (const float*)d_in[8];
  const float* bf2 = (const float*)d_in[9];

  const int N = in_sizes[0] / 256;   // IN_DIM = 256
  const int E = in_sizes[1] / 2;     // edge_index is [2, E]
  const int* src = ei;
  const int* dst = ei + E;

  char*  ws   = (char*)d_ws;
  float* dinv = (float*)ws;
  size_t offA = (((size_t)N * 4) + 255) & ~(size_t)255;
  float* bufA = (float*)(ws + offA);
  float* bufB = (float*)(ws + offA + (size_t)N * 128 * 4);
  float* out  = (float*)d_out;

  dim3 blk(NTHREADS);
  int gN    = (N + NTHREADS - 1) / NTHREADS;
  int gE    = (E + NTHREADS - 1) / NTHREADS;
  int gGemm = (N + 127) / 128;
  int gInit = (int)(((size_t)N * 32 + NTHREADS - 1) / NTHREADS);
  int gScat = (int)(((size_t)E * 32 + NTHREADS - 1) / NTHREADS);

  // Normalization coefficients.
  deg_init_k<<<gN, blk, 0, stream>>>(dinv, N);
  deg_count_k<<<gE, blk, 0, stream>>>(dst, dinv, E);
  deg_rsqrt_k<<<gN, blk, 0, stream>>>(dinv, N);

  // Layer 1: T1 = x @ W1 ; agg1 = scatter(T1) ; (b1 + relu fused into next GEMM)
  gemm_wmma_k<256, 128, false, false><<<gGemm, blk, 0, stream>>>(x, W1, nullptr, nullptr, bufA, N);
  selfloop_init_k<<<gInit, blk, 0, stream>>>(bufA, dinv, bufB, N);
  scatter_k<<<gScat, blk, 0, stream>>>(bufA, dinv, src, dst, bufB, E);

  // Layer 2: T2 = relu(agg1 + b1) @ W2 ; agg2 = scatter(T2)
  gemm_wmma_k<128, 128, true, false><<<gGemm, blk, 0, stream>>>(bufB, W2, b1, nullptr, bufA, N);
  selfloop_init_k<<<gInit, blk, 0, stream>>>(bufA, dinv, bufB, N);
  scatter_k<<<gScat, blk, 0, stream>>>(bufA, dinv, src, dst, bufB, E);

  // MLP head: h3 = relu( relu(agg2 + b2) @ Wf1 + bf1 )
  gemm_wmma_k<128, 128, true, true><<<gGemm, blk, 0, stream>>>(bufB, Wf1, b2, bf1, bufA, N);
  // out = h3 @ Wf2 + bf2
  gemm_wmma_k<128, 64, false, false><<<gGemm, blk, 0, stream>>>(bufA, Wf2, nullptr, bf2, out, N);
}
